// MoELayer_43121471652238
// MI455X (gfx1250) — compile-verified, matrix operand order checked by reading, not run on hardware
//
#include <hip/hip_runtime.h>

// ---------------- problem constants ----------------
constexpr int B_ = 4, T_ = 2048, C_ = 1024, F_ = 4096, E_ = 8, K_ = 2;
constexpr int NTOK    = B_ * T_;        // 8192 tokens
constexpr int NASSIGN = NTOK * K_;      // 16384 token-expert assignments (exact)
constexpr int MT      = 128;            // GEMM M tile
constexpr int ACAP    = NASSIGN + E_ * MT; // padded capacity (>= worst-case padded total)
constexpr int GRID_M  = ACAP / MT;      // 136 max M tiles

// LDS row stride (u16 elements): 80B rows, uint2-store friendly, conflict-benign
constexpr int AS = 40;

typedef __attribute__((ext_vector_type(16))) __bf16 v16bf;
typedef __attribute__((ext_vector_type(8)))  float  v8f;

union Frag { unsigned u[8]; v16bf v; };

__device__ __forceinline__ unsigned short f2bf(float f) {
  unsigned u = __float_as_uint(f);
  u += 0x7FFFu + ((u >> 16) & 1u);       // round-to-nearest-even
  return (unsigned short)(u >> 16);
}

// ---------------- init: zero counts + padded assignment list ----------------
__global__ __launch_bounds__(256) void init_kernel(int* meta, int* assign_tok) {
  int i = blockIdx.x * 256 + threadIdx.x;
  if (i < 8) meta[i] = 0;                 // counts[8]
  if (i < ACAP) assign_tok[i] = 0;        // padded rows gather token 0 (never combined)
}

// ---------------- f32 -> bf16 (RNE), linear ----------------
__global__ __launch_bounds__(256) void convert_kernel(const float* __restrict__ src,
                                                      unsigned short* __restrict__ dst,
                                                      long n) {
  long i = (long)blockIdx.x * 256 + threadIdx.x;
  long stride = (long)gridDim.x * 256;
  for (; i < n; i += stride) dst[i] = f2bf(src[i]);
}

// ---------------- f32 [E][KD][ND] -> bf16 [E][ND][KD] (tiled transpose) ----------------
__global__ __launch_bounds__(256) void transpose_convert_kernel(
    const float* __restrict__ src, unsigned short* __restrict__ dst, int KD, int ND) {
  __shared__ unsigned short tile[32][33];
  const int e  = blockIdx.z;
  const int k0 = blockIdx.y * 32;
  const int n0 = blockIdx.x * 32;
  const int tid = threadIdx.x;
  const float* S = src + ((size_t)e * KD + k0) * ND + n0;
#pragma unroll
  for (int j = 0; j < 4; ++j) {          // read coalesced along N
    int flat = tid + j * 256;
    int r = flat >> 5, c = flat & 31;
    tile[r][c] = f2bf(S[(size_t)r * ND + c]);
  }
  __syncthreads();
  unsigned short* D = dst + ((size_t)e * ND + n0) * KD + k0;
#pragma unroll
  for (int j = 0; j < 4; ++j) {          // write coalesced along K
    int flat = tid + j * 256;
    int rn = flat >> 5, ck = flat & 31;
    D[(size_t)rn * KD + ck] = tile[ck][rn];
  }
}

// ---------------- gating: one wave32 per token ----------------
__global__ __launch_bounds__(256) void gating_kernel(const float* __restrict__ x,
                                                     const float* __restrict__ Wg,
                                                     int* __restrict__ counts,
                                                     int* __restrict__ topk_idx,
                                                     float* __restrict__ topk_w) {
  int t = blockIdx.x * 8 + (threadIdx.x >> 5);
  int lane = threadIdx.x & 31;
  if (t >= NTOK) return;
  float lg[E_];
#pragma unroll
  for (int e = 0; e < E_; ++e) lg[e] = 0.f;
  const float* xr = x + (size_t)t * C_;
  for (int c = lane; c < C_; c += 32) {
    float xv = xr[c];
    const float* w = Wg + (size_t)c * E_;
#pragma unroll
    for (int e = 0; e < E_; ++e) lg[e] += xv * w[e];
  }
#pragma unroll
  for (int e = 0; e < E_; ++e)
#pragma unroll
    for (int off = 16; off > 0; off >>= 1) lg[e] += __shfl_xor(lg[e], off, 32);
  if (lane == 0) {
    float m = lg[0];
#pragma unroll
    for (int e = 1; e < E_; ++e) m = fmaxf(m, lg[e]);
    float p[E_];
#pragma unroll
    for (int e = 0; e < E_; ++e) p[e] = expf(lg[e] - m);
    int i0 = 0; float p0 = p[0];
#pragma unroll
    for (int e = 1; e < E_; ++e) if (p[e] > p0) { p0 = p[e]; i0 = e; }
    int i1 = (i0 == 0) ? 1 : 0; float p1 = p[i1];
#pragma unroll
    for (int e = 0; e < E_; ++e) if (e != i0 && p[e] > p1) { p1 = p[e]; i1 = e; }
    float s = p0 + p1;
    topk_idx[t * 2 + 0] = i0; topk_idx[t * 2 + 1] = i1;
    topk_w[t * 2 + 0] = p0 / s; topk_w[t * 2 + 1] = p1 / s;
    atomicAdd(&counts[i0], 1);
    atomicAdd(&counts[i1], 1);
  }
}

// ---------------- prefix: padded per-expert bases + cursors ----------------
__global__ void prefix_kernel(int* meta) {
  int* counts  = meta;
  int* cursors = meta + 8;
  int* base    = meta + 16;     // base[0..8]
  int acc = 0;
  base[0] = 0;
  for (int e = 0; e < E_; ++e) {
    cursors[e] = acc;
    acc += (counts[e] + (MT - 1)) & ~(MT - 1);
    base[e + 1] = acc;
  }
}

// ---------------- scatter: compacted per-expert lists ----------------
__global__ __launch_bounds__(256) void scatter_kernel(const int* __restrict__ topk_idx,
                                                      int* __restrict__ cursors,
                                                      int* __restrict__ assign_tok,
                                                      int* __restrict__ topk_slot) {
  int t = blockIdx.x * 256 + threadIdx.x;
  if (t >= NTOK) return;
#pragma unroll
  for (int k = 0; k < K_; ++k) {
    int e = topk_idx[t * 2 + k];
    int slot = atomicAdd(&cursors[e], 1);
    assign_tok[slot] = t;
    topk_slot[t * 2 + k] = slot;
  }
}

// ---------------- routed tiled bf16 WMMA GEMM (weights pre-transposed) ----------------
// PHASE1: h = gelu(gather(xb) @ W1[e] + b1[e])  -> bf16 Hout [slot][F]
// PHASE2: y = Hout @ W2[e] + b2[e]              -> f32  Yout [slot][C]
// Wt is [E][NDIM][KDIM] bf16, so B fragments are K-contiguous like A fragments.
template <int KDIM, int NDIM, bool PHASE1>
__global__ __launch_bounds__(256) void moe_gemm_kernel(
    const unsigned short* __restrict__ Abase,   // xb (PHASE1) or h_buf (PHASE2)
    const int* __restrict__ assign_tok,
    const unsigned short* __restrict__ Wt,      // [E][NDIM][KDIM] bf16 (pre-transposed)
    const float* __restrict__ bias,             // [E][NDIM]
    const int* __restrict__ base,               // base[0..8]
    unsigned short* __restrict__ Hout,
    float* __restrict__ Yout) {
  __shared__ __align__(16) unsigned short As[128 * AS];
  __shared__ __align__(16) unsigned short Bs[128 * AS];

  const int m0 = blockIdx.x * MT;
  if (m0 >= base[8]) return;                    // beyond padded assignment total
  int e = 0;
  while (e < 7 && m0 >= base[e + 1]) ++e;       // tile's expert (tiles never straddle)

  const int n0   = blockIdx.y * 128;
  const int tid  = threadIdx.x;
  const int lane = tid & 31;
  const int wid  = tid >> 5;
  const int wr   = wid >> 1;                    // wave row 0..3 (32 rows each)
  const int wc   = wid & 1;                     // wave col 0..1 (64 cols each)

  const unsigned short* We = Wt + (size_t)e * NDIM * KDIM;

  // per-thread staging slices: 2 rows-of-8-bf16 each for A and B (fixed across K)
  const unsigned short* aptr[2];
  const unsigned short* bptr[2];
  int srow[2], sq[2];
#pragma unroll
  for (int j = 0; j < 2; ++j) {
    int flat = tid + j * 256;
    int r = flat >> 2, q = flat & 3;
    srow[j] = r; sq[j] = q;
    int slot = m0 + r;
    int arow = PHASE1 ? assign_tok[slot] : slot;
    aptr[j] = Abase + (size_t)arow * KDIM + q * 8;
    bptr[j] = We + (size_t)(n0 + r) * KDIM + q * 8;
  }

  v8f acc[2][4];
  const v8f vzero = {0.f, 0.f, 0.f, 0.f, 0.f, 0.f, 0.f, 0.f};
#pragma unroll
  for (int mi = 0; mi < 2; ++mi)
#pragma unroll
    for (int ni = 0; ni < 4; ++ni) acc[mi][ni] = vzero;

  // prologue: fetch K-tile 0 into registers, park in LDS
  uint4 abuf[2], bbuf[2];
#pragma unroll
  for (int j = 0; j < 2; ++j) {
    abuf[j] = *(const uint4*)(aptr[j]);
    bbuf[j] = *(const uint4*)(bptr[j]);
  }
#pragma unroll
  for (int j = 0; j < 2; ++j) {
    unsigned short* da = &As[srow[j] * AS + sq[j] * 8];
    ((uint2*)da)[0] = make_uint2(abuf[j].x, abuf[j].y);
    ((uint2*)da)[1] = make_uint2(abuf[j].z, abuf[j].w);
    unsigned short* db = &Bs[srow[j] * AS + sq[j] * 8];
    ((uint2*)db)[0] = make_uint2(bbuf[j].x, bbuf[j].y);
    ((uint2*)db)[1] = make_uint2(bbuf[j].z, bbuf[j].w);
  }
  __syncthreads();

  for (int kt = 0;;) {
    const int kn = kt + 32;
    // software pipeline: pull next K-tile into registers while WMMAs run
    if (kn < KDIM) {
#pragma unroll
      for (int j = 0; j < 2; ++j) {
        abuf[j] = *(const uint4*)(aptr[j] + kn);
        bbuf[j] = *(const uint4*)(bptr[j] + kn);
        if (kn + 32 < KDIM) __builtin_prefetch(bptr[j] + kn + 32, 0, 1);
      }
    }

    // fragments: 8x ds_load_b32 each (ISA 16-bit layout: K pairs/VGPR, lane>=16 -> +8)
    Frag afr[2], bfr[4];
    const int kb = (lane & 16) ? 8 : 0;
#pragma unroll
    for (int mi = 0; mi < 2; ++mi) {
      int r = wr * 32 + mi * 16 + (lane & 15);
#pragma unroll
      for (int vi = 0; vi < 8; ++vi) {
        int kk = ((vi < 4) ? 2 * vi : 16 + 2 * (vi - 4)) + kb;
        afr[mi].u[vi] = *(const unsigned*)&As[r * AS + kk];
      }
    }
#pragma unroll
    for (int ni = 0; ni < 4; ++ni) {
      int n = wc * 64 + ni * 16 + (lane & 15);
#pragma unroll
      for (int vi = 0; vi < 8; ++vi) {
        int kk = ((vi < 4) ? 2 * vi : 16 + 2 * (vi - 4)) + kb;
        bfr[ni].u[vi] = *(const unsigned*)&Bs[n * AS + kk];
      }
    }
    // 8x V_WMMA_F32_16X16X32_BF16 per wave per K-step
#pragma unroll
    for (int mi = 0; mi < 2; ++mi)
#pragma unroll
      for (int ni = 0; ni < 4; ++ni)
        acc[mi][ni] = __builtin_amdgcn_wmma_f32_16x16x32_bf16(
            false, afr[mi].v, false, bfr[ni].v, (short)0, acc[mi][ni], false, false);

    if (kn >= KDIM) break;
    __syncthreads();                            // everyone done reading LDS
#pragma unroll
    for (int j = 0; j < 2; ++j) {
      unsigned short* da = &As[srow[j] * AS + sq[j] * 8];
      ((uint2*)da)[0] = make_uint2(abuf[j].x, abuf[j].y);
      ((uint2*)da)[1] = make_uint2(abuf[j].z, abuf[j].w);
      unsigned short* db = &Bs[srow[j] * AS + sq[j] * 8];
      ((uint2*)db)[0] = make_uint2(bbuf[j].x, bbuf[j].y);
      ((uint2*)db)[1] = make_uint2(bbuf[j].z, bbuf[j].w);
    }
    __syncthreads();
    kt = kn;
  }

  // --- epilogue: bias (+ exact-erf GELU -> bf16) or f32 store ---
  const float* bp = bias + (size_t)e * NDIM;
#pragma unroll
  for (int mi = 0; mi < 2; ++mi) {
#pragma unroll
    for (int ni = 0; ni < 4; ++ni) {
      int n = n0 + wc * 64 + ni * 16 + (lane & 15);
      float bv = bp[n];
#pragma unroll
      for (int i = 0; i < 8; ++i) {
        int row = wr * 32 + mi * 16 + i + ((lane & 16) ? 8 : 0);
        size_t slot = (size_t)(m0 + row);
        float v = acc[mi][ni][i] + bv;
        if (PHASE1) {
          v = 0.5f * v * (1.0f + erff(v * 0.70710678118654752f));
          Hout[slot * NDIM + n] = f2bf(v);
        } else {
          Yout[slot * NDIM + n] = v;
        }
      }
    }
  }
}

// ---------------- combine: out[t] = w0*y[s0] + w1*y[s1] ----------------
__global__ __launch_bounds__(256) void combine_kernel(const float* __restrict__ y,
                                                      const float* __restrict__ topk_w,
                                                      const int* __restrict__ topk_slot,
                                                      float* __restrict__ out) {
  int i = blockIdx.x * 256 + threadIdx.x;     // over NTOK*C = 8.4M
  int t = i >> 10;
  int c = i & (C_ - 1);
  int s0 = topk_slot[t * 2 + 0], s1 = topk_slot[t * 2 + 1];
  float w0 = topk_w[t * 2 + 0],  w1 = topk_w[t * 2 + 1];
  out[i] = w0 * y[(size_t)s0 * C_ + c] + w1 * y[(size_t)s1 * C_ + c];
}

// ---------------- host launcher ----------------
extern "C" void kernel_launch(void* const* d_in, const int* in_sizes, int n_in,
                              void* d_out, int out_size, void* d_ws, size_t ws_size,
                              hipStream_t stream) {
  const float* x  = (const float*)d_in[0];
  const float* Wg = (const float*)d_in[1];
  const float* W1 = (const float*)d_in[2];
  const float* b1 = (const float*)d_in[3];
  const float* W2 = (const float*)d_in[4];
  const float* b2 = (const float*)d_in[5];
  float* out = (float*)d_out;

  char* ws = (char*)d_ws;
  size_t off = 0;
  auto carve = [&](size_t bytes) {
    size_t o = off;
    off = (off + bytes + 255) & ~(size_t)255;
    return o;
  };
  int*            meta       = (int*)           (ws + carve(256));                       // counts[8],cursors[8],base[9]
  int*            topk_idx   = (int*)           (ws + carve(sizeof(int)   * NASSIGN));
  float*          topk_w     = (float*)         (ws + carve(sizeof(float) * NASSIGN));
  int*            topk_slot  = (int*)           (ws + carve(sizeof(int)   * NASSIGN));
  int*            assign_tok = (int*)           (ws + carve(sizeof(int)   * ACAP));
  unsigned short* xb         = (unsigned short*)(ws + carve(sizeof(short) * (size_t)NTOK * C_));
  unsigned short* W1t        = (unsigned short*)(ws + carve(sizeof(short) * (size_t)E_ * C_ * F_)); // [E][F][C]
  unsigned short* W2t        = (unsigned short*)(ws + carve(sizeof(short) * (size_t)E_ * F_ * C_)); // [E][C][F]
  unsigned short* h_buf      = (unsigned short*)(ws + carve(sizeof(short) * (size_t)ACAP * F_));
  float*          y_buf      = (float*)         (ws + carve(sizeof(float) * (size_t)ACAP * C_));
  (void)ws_size; (void)in_sizes; (void)n_in; (void)out_size;

  int* counts  = meta;
  int* cursors = meta + 8;
  int* base    = meta + 16;

  // 1. init
  init_kernel<<<(ACAP + 255) / 256, 256, 0, stream>>>(meta, assign_tok);
  // 2. bf16 conversions: x linear; W1/W2 transposed to [E][N][K]
  convert_kernel<<<4096, 256, 0, stream>>>(x, xb, (long)NTOK * C_);
  transpose_convert_kernel<<<dim3(F_ / 32, C_ / 32, E_), 256, 0, stream>>>(W1, W1t, C_, F_);
  transpose_convert_kernel<<<dim3(C_ / 32, F_ / 32, E_), 256, 0, stream>>>(W2, W2t, F_, C_);
  // 3. gating (f32, one wave per token)
  gating_kernel<<<NTOK / 8, 256, 0, stream>>>(x, Wg, counts, topk_idx, topk_w);
  // 4. padded prefix
  prefix_kernel<<<1, 1, 0, stream>>>(meta);
  // 5. scatter compacted assignments
  scatter_kernel<<<NTOK / 256, 256, 0, stream>>>(topk_idx, cursors, assign_tok, topk_slot);
  // 6. GEMM1: h = gelu(x @ W1 + b1)   (K=1024, N=4096)
  moe_gemm_kernel<C_, F_, true><<<dim3(GRID_M, F_ / 128), 256, 0, stream>>>(
      xb, assign_tok, W1t, b1, base, h_buf, (float*)nullptr);
  // 7. GEMM2: y = h @ W2 + b2         (K=4096, N=1024)
  moe_gemm_kernel<F_, C_, false><<<dim3(GRID_M, C_ / 128), 256, 0, stream>>>(
      h_buf, assign_tok, W2t, b2, base, (unsigned short*)nullptr, y_buf);
  // 8. combine
  combine_kernel<<<(NTOK * C_) / 256, 256, 0, stream>>>(y_buf, topk_w, topk_slot, out);
}
